// FluorescenceLoss_59236188946647
// MI455X (gfx1250) — matrix operand order repaired and checked by reading, not compiled, not run on hardware
//
#include <hip/hip_runtime.h>
#include <math.h>

// ---------------------------------------------------------------------------
// FluorescenceLoss for MI455X (gfx1250, wave32).
// SSIM Gaussian filtering is mapped to v_wmma_f32_16x16x32_f16 as a banded
// GEMM (11-tap band fits in K=32). Everything else is bandwidth-trivial.
// ---------------------------------------------------------------------------

#define BATCH 32
#define H0IMG 512

// accumulator layout (float indices into d_ws)
#define IDX_L1   0
#define IDX_MSE  1
#define IDX_CS   2            // 5 scales * 32 batch
#define IDX_SS   (2 + 160)    // 5 scales * 32 batch
#define IDX_GRAD (2 + 320)    // 5 scales
#define IDX_FREQ (2 + 325)
#define ACC_FLOATS 512

typedef __attribute__((ext_vector_type(16))) _Float16 v16h;
typedef __attribute__((ext_vector_type(8)))  float    v8f;

union V16H { v16h v; _Float16 e[16]; };
union V8F  { v8f  v; float    e[8];  };

__device__ __forceinline__ float warpSum(float v) {
    #pragma unroll
    for (int o = 16; o > 0; o >>= 1) v += __shfl_down(v, o, 32);
    return v;
}

__device__ __forceinline__ float blockSum(float v) {
    __shared__ float sw[32];
    int lane = threadIdx.x & 31, wid = threadIdx.x >> 5;
    v = warpSum(v);
    __syncthreads();
    if (lane == 0) sw[wid] = v;
    __syncthreads();
    float r = 0.f;
    if (wid == 0) {
        int nw = ((int)blockDim.x + 31) >> 5;
        r = (lane < nw) ? sw[lane] : 0.f;
        r = warpSum(r);
    }
    return r;  // valid in thread 0
}

// normalized 11-tap Gaussian, sigma=1.5  (2*sigma^2 = 4.5)
__device__ __forceinline__ void gauss11(float g[11]) {
    float s = 0.f;
    #pragma unroll
    for (int i = 0; i < 11; i++) { float c = (float)i - 5.f; g[i] = __expf(-c * c / 4.5f); s += g[i]; }
    float inv = 1.f / s;
    #pragma unroll
    for (int i = 0; i < 11; i++) g[i] *= inv;
}

// symmetric-pad index for DWT: xe[m], m in [0, N+5)
__device__ __forceinline__ int symIdx(int m, int N) {
    if (m == 0) return 1;
    if (m == 1) return 0;
    if (m < N + 2) return m - 2;
    if (m == N + 2) return N - 1;
    if (m == N + 3) return N - 2;
    return N - 3;
}

// ---------------------------------------------------------------------------
__global__ void k_zero(float* __restrict__ acc) {
    if (threadIdx.x < ACC_FLOATS) acc[threadIdx.x] = 0.f;
}

// d = p - t written to D; sum|d|, sum d^2 accumulated
__global__ void k_diff(const float* __restrict__ p, const float* __restrict__ t,
                       float* __restrict__ D, float* __restrict__ acc, size_t n) {
    size_t stride = (size_t)gridDim.x * blockDim.x;
    float s1 = 0.f, s2 = 0.f;
    for (size_t i = (size_t)blockIdx.x * blockDim.x + threadIdx.x; i < n; i += stride) {
        if (i + 4 * stride < n) {
            __builtin_prefetch(p + i + 4 * stride, 0, 1);   // global_prefetch_b8
            __builtin_prefetch(t + i + 4 * stride, 0, 1);
        }
        float d = p[i] - t[i];
        D[i] = d;
        s1 += fabsf(d);
        s2 += d * d;
    }
    s1 = blockSum(s1);
    s2 = blockSum(s2);
    if (threadIdx.x == 0) { atomicAdd(&acc[IDX_L1], s1); atomicAdd(&acc[IDX_MSE], s2); }
}

// 2x2 average pool, stride 2 (BATCH planes of Hin x Win)
__global__ void k_pool2(const float* __restrict__ in, float* __restrict__ out, int Hin, int Win) {
    int Ho = Hin >> 1, Wo = Win >> 1;
    size_t n = (size_t)BATCH * Ho * Wo;
    size_t stride = (size_t)gridDim.x * blockDim.x;
    for (size_t idx = (size_t)blockIdx.x * blockDim.x + threadIdx.x; idx < n; idx += stride) {
        int j = (int)(idx % Wo);
        size_t r = idx / Wo;
        int i = (int)(r % Ho);
        int b = (int)(r / Ho);
        const float* ip = in + ((size_t)b * Hin + 2 * i) * Win + 2 * j;
        out[idx] = 0.25f * (ip[0] + ip[1] + ip[Win] + ip[Win + 1]);
    }
}

// ---------------------------------------------------------------------------
// Horizontal 11-tap valid conv as WMMA banded GEMM, fused products.
// One wave -> one 16x16 output tile for all 5 planes (X, Y, XX, YY, XY).
// out[m][n] = sum_K A[m][K] * G[K][n],  A[m][K] = in[r0+m][c0+K],
// G[K][n] = g[K-n] for 0<=K-n<=10 (band; rows 26..31 naturally zero).
// T layout: plane q in {mu1src,mu2src,xx,yy,xy}: T[(q*BATCH+b)*H + r]*Wout + c
// ---------------------------------------------------------------------------
__global__ void k_ssim_h(const float* __restrict__ X, const float* __restrict__ Y,
                         float* __restrict__ T, int H, int W) {
    const int Wout = W - 10;
    const int ct = (Wout + 15) >> 4;
    const int rt = H >> 4;
    int wave = ((int)(blockIdx.x * blockDim.x + threadIdx.x)) >> 5;
    int lane = threadIdx.x & 31;
    int total = ct * rt * BATCH;
    if (wave >= total) return;  // wave-uniform: EXEC stays all-ones for WMMA

    int c0 = (wave % ct) << 4;
    int tmp = wave / ct;
    int r0 = (tmp % rt) << 4;
    int b = tmp / rt;

    float g[11];
    gauss11(g);

    int m    = lane & 15;           // A row / B col / D col
    int koff = (lane >> 4) << 3;    // A K-half select
    const float* xr = X + ((size_t)b * H + (r0 + m)) * W;
    const float* yr = Y + ((size_t)b * H + (r0 + m)) * W;

    V16H A[5];
    #pragma unroll
    for (int h = 0; h < 16; h++) {
        int p = h >> 1;
        int K = (koff << 1) + h; (void)p;
        int col = c0 + K;
        if (col > W - 1) col = W - 1;   // value only hits zero band entries
        float xv = xr[col], yv = yr[col];
        A[0].e[h] = (_Float16)xv;
        A[1].e[h] = (_Float16)yv;
        A[2].e[h] = (_Float16)(xv * xv);
        A[3].e[h] = (_Float16)(yv * yv);
        A[4].e[h] = (_Float16)(xv * yv);
    }

    int kb = (lane >> 4) << 4;      // B K-half select
    V16H bm;
    #pragma unroll
    for (int h = 0; h < 16; h++) {
        int d = kb + h - m;         // K - n
        bm.e[h] = (d >= 0 && d <= 10) ? (_Float16)g[d] : (_Float16)0.f;
    }

    V8F acc[5];
    #pragma unroll
    for (int q = 0; q < 5; q++) {
        v8f z = {};
        acc[q].v = __builtin_amdgcn_wmma_f32_16x16x32_f16(
            false, A[q].v, false, bm.v, (short)0, z, false, false);
    }

    int colOut = c0 + m;
    int rbase  = r0 + ((lane >> 4) << 3);
    if (colOut < Wout) {
        #pragma unroll
        for (int q = 0; q < 5; q++) {
            float* op = T + ((size_t)(q * BATCH + b)) * H * Wout;
            #pragma unroll
            for (int r = 0; r < 8; r++)
                op[(size_t)(rbase + r) * Wout + colOut] = acc[q].e[r];
        }
    }
}

// ---------------------------------------------------------------------------
// Vertical 11-tap valid conv as WMMA banded GEMM, fused SSIM/CS reduction.
// out[m][n] = sum_K G[m][K] * B[K][n],  G[m][K] = g[K-m], B[K][n]=T[r0+K][c0+n]
// ---------------------------------------------------------------------------
__global__ void k_ssim_v(const float* __restrict__ T, float* __restrict__ acc,
                         int H, int Wout, int scale) {
    const int Hout = H - 10;
    const int ct = (Wout + 15) >> 4;
    const int rt = (Hout + 15) >> 4;
    int wave = ((int)(blockIdx.x * blockDim.x + threadIdx.x)) >> 5;
    int lane = threadIdx.x & 31;
    int total = ct * rt * BATCH;
    if (wave >= total) return;

    int c0 = (wave % ct) << 4;
    int tmp = wave / ct;
    int r0 = (tmp % rt) << 4;
    int b = tmp / rt;

    float g[11];
    gauss11(g);

    int m    = lane & 15;
    int koff = (lane >> 4) << 3;
    V16H am;
    #pragma unroll
    for (int h = 0; h < 16; h++) {
        int p = h >> 1;
        int K = (koff << 1) + h; (void)p;
        int d = K - m;
        am.e[h] = (d >= 0 && d <= 10) ? (_Float16)g[d] : (_Float16)0.f;
    }

    int kb = (lane >> 4) << 4;
    int colIn = c0 + m; if (colIn > Wout - 1) colIn = Wout - 1;

    V8F res[5];
    #pragma unroll
    for (int q = 0; q < 5; q++) {
        const float* tp = T + ((size_t)(q * BATCH + b)) * H * Wout;
        V16H bm;
        #pragma unroll
        for (int h = 0; h < 16; h++) {
            int row = r0 + kb + h;
            if (row > H - 1) row = H - 1;  // only feeds zero band entries
            bm.e[h] = (_Float16)tp[(size_t)row * Wout + colIn];
        }
        v8f z = {};
        res[q].v = __builtin_amdgcn_wmma_f32_16x16x32_f16(
            false, am.v, false, bm.v, (short)0, z, false, false);
    }

    const float C1 = 1e-4f, C2 = 9e-4f;
    float csS = 0.f, ssS = 0.f;
    int rbase = r0 + ((lane >> 4) << 3);
    int col = c0 + m;
    #pragma unroll
    for (int rr = 0; rr < 8; rr++) {
        int row = rbase + rr;
        if (row < Hout && col < Wout) {
            float mu1 = res[0].e[rr], mu2 = res[1].e[rr];
            float s1  = res[2].e[rr] - mu1 * mu1;
            float s2  = res[3].e[rr] - mu2 * mu2;
            float s12 = res[4].e[rr] - mu1 * mu2;
            float cs  = (2.f * s12 + C2) / (s1 + s2 + C2);
            float ss  = (2.f * mu1 * mu2 + C1) / (mu1 * mu1 + mu2 * mu2 + C1) * cs;
            csS += cs;
            ssS += ss;
        }
    }
    csS = warpSum(csS);
    ssS = warpSum(ssS);
    if (lane == 0) {
        atomicAdd(&acc[IDX_CS + scale * BATCH + b], csS);
        atomicAdd(&acc[IDX_SS + scale * BATCH + b], ssS);
    }
}

// ---------------------------------------------------------------------------
// Gradient loss at one scale: SAME-padded conv of d with gauss(size) x [-1,0,1]
// (and transpose), sum sqrt(c^2 + 1e-6).
// ---------------------------------------------------------------------------
__global__ void k_grad(const float* __restrict__ D, int Hs, int Ws, int s,
                       float* __restrict__ acc) {
    int size = 3 + 2 * s;
    int ph = (size - 1) >> 1;
    __shared__ float g[11];
    __shared__ float inv2S;
    if ((int)threadIdx.x < size) {
        float c = (float)threadIdx.x - 0.5f * (float)(size - 1);
        float sig = (float)size / 3.f;
        g[threadIdx.x] = __expf(-c * c / (2.f * sig * sig));
    }
    __syncthreads();
    if (threadIdx.x == 0) {
        float S = 0.f;
        for (int i = 0; i < size; i++) S += g[i];
        inv2S = 0.5f / S;
    }
    __syncthreads();

    size_t n = (size_t)BATCH * Hs * Ws;
    size_t stride = (size_t)gridDim.x * blockDim.x;
    float loc = 0.f;
    for (size_t idx = (size_t)blockIdx.x * blockDim.x + threadIdx.x; idx < n; idx += stride) {
        int j = (int)(idx % Ws);
        size_t r2 = idx / Ws;
        int i = (int)(r2 % Hs);
        int b = (int)(r2 / Hs);
        const float* P = D + (size_t)b * Hs * Ws;
        float cx = 0.f, cy = 0.f;
        for (int a = 0; a < size; a++) {
            int ii = i + a - ph;
            if (ii >= 0 && ii < Hs) {
                float rightv = (j + 1 < Ws) ? P[(size_t)ii * Ws + j + 1] : 0.f;
                float leftv  = (j - 1 >= 0) ? P[(size_t)ii * Ws + j - 1] : 0.f;
                cx += g[a] * (rightv - leftv);
            }
            int jj = j + a - ph;
            if (jj >= 0 && jj < Ws) {
                float dn = (i + 1 < Hs) ? P[(size_t)(i + 1) * Ws + jj] : 0.f;
                float up = (i - 1 >= 0) ? P[(size_t)(i - 1) * Ws + jj] : 0.f;
                cy += g[a] * (dn - up);
            }
        }
        cx *= inv2S;
        cy *= inv2S;
        loc += sqrtf(cx * cx + 1e-6f) + sqrtf(cy * cy + 1e-6f);
    }
    loc = blockSum(loc);
    if (threadIdx.x == 0) atomicAdd(&acc[IDX_GRAD + s], loc);
}

// ---------------------------------------------------------------------------
// db2 DWT stage 1 (along H): D(32,512,512) -> LO,HI (32,257,512)
// ---------------------------------------------------------------------------
__global__ void k_dwt_v(const float* __restrict__ D, float* __restrict__ LO,
                        float* __restrict__ HI) {
    const int H = 512, W = 512, HO = 257;
    const float rlo[4] = {0.48296291314469025f, 0.8365163037378079f,
                          0.22414386804185735f, -0.12940952255092145f};
    const float rhi[4] = {0.12940952255092145f, -0.22414386804185735f,
                          0.8365163037378079f, -0.48296291314469025f};
    size_t n = (size_t)BATCH * HO * W;
    size_t stride = (size_t)gridDim.x * blockDim.x;
    for (size_t idx = (size_t)blockIdx.x * blockDim.x + threadIdx.x; idx < n; idx += stride) {
        int w = (int)(idx % W);
        size_t r = idx / W;
        int io = (int)(r % HO);
        int b = (int)(r / HO);
        const float* P = D + (size_t)b * H * W;
        float lo = 0.f, hi = 0.f;
        #pragma unroll
        for (int t = 0; t < 4; t++) {
            int ii = symIdx(2 * io + t, H);
            float v = P[(size_t)ii * W + w];
            lo += rlo[t] * v;
            hi += rhi[t] * v;
        }
        LO[idx] = lo;
        HI[idx] = hi;
    }
}

// stage 2 (along W) fused with Huber reduction over cH,cV,cD
__global__ void k_dwt_h(const float* __restrict__ LO, const float* __restrict__ HI,
                        float* __restrict__ acc) {
    const int W = 512, HO = 257, WO = 257;
    const float rlo[4] = {0.48296291314469025f, 0.8365163037378079f,
                          0.22414386804185735f, -0.12940952255092145f};
    const float rhi[4] = {0.12940952255092145f, -0.22414386804185735f,
                          0.8365163037378079f, -0.48296291314469025f};
    size_t n = (size_t)BATCH * HO * WO;
    size_t stride = (size_t)gridDim.x * blockDim.x;
    float loc = 0.f;
    for (size_t idx = (size_t)blockIdx.x * blockDim.x + threadIdx.x; idx < n; idx += stride) {
        int jo = (int)(idx % WO);
        size_t r = idx / WO;
        int io = (int)(r % HO);
        int b = (int)(r / HO);
        const float* lr = LO + ((size_t)b * HO + io) * W;
        const float* hr = HI + ((size_t)b * HO + io) * W;
        float cH = 0.f, cD = 0.f, cV = 0.f;
        #pragma unroll
        for (int t = 0; t < 4; t++) {
            int jj = symIdx(2 * jo + t, W);
            float hv = hr[jj], lv = lr[jj];
            cH += rlo[t] * hv;
            cD += rhi[t] * hv;
            cV += rhi[t] * lv;
        }
        float a;
        a = fabsf(cH); loc += (a < 1.f) ? 0.5f * cH * cH : a - 0.5f;
        a = fabsf(cV); loc += (a < 1.f) ? 0.5f * cV * cV : a - 0.5f;
        a = fabsf(cD); loc += (a < 1.f) ? 0.5f * cD * cD : a - 0.5f;
    }
    loc = blockSum(loc);
    if (threadIdx.x == 0) atomicAdd(&acc[IDX_FREQ], loc);
}

// ---------------------------------------------------------------------------
__global__ void k_final(const float* __restrict__ acc, const float* __restrict__ lw,
                        const float* __restrict__ temp, float* __restrict__ out) {
    if (threadIdx.x != 0 || blockIdx.x != 0) return;
    float T = temp[0];
    if (T < 0.1f) T = 0.1f;
    float wv[6];
    float mx = -1e30f;
    for (int i = 0; i < 6; i++) { wv[i] = lw[i] / T; if (wv[i] > mx) mx = wv[i]; }
    float se = 0.f;
    for (int i = 0; i < 6; i++) { wv[i] = __expf(wv[i] - mx); se += wv[i]; }
    for (int i = 0; i < 6; i++) wv[i] /= se;

    const float wms[5] = {0.0448f, 0.2856f, 0.3001f, 0.2363f, 0.1333f};
    float ms = 0.f;
    for (int b = 0; b < BATCH; b++) {
        float p = 1.f;
        for (int i = 0; i < 5; i++) {
            int Hs = H0IMG >> i;
            float Hv = (float)(Hs - 10);
            float denom = Hv * Hv;
            float v = ((i < 4) ? acc[IDX_CS + i * BATCH + b]
                               : acc[IDX_SS + 4 * BATCH + b]) / denom;
            v = (v > 0.f) ? v : 0.f;
            p *= powf(v, wms[i]);
        }
        ms += p;
    }
    ms *= (1.f / (float)BATCH);

    float gl = 0.f;
    for (int s = 0; s < 5; s++) {
        int Hs = H0IMG >> s;
        gl += (1.f / (float)(1 << s)) * acc[IDX_GRAD + s] /
              (2.f * (float)BATCH * (float)Hs * (float)Hs);
    }
    gl *= 0.2f;

    float fl = acc[IDX_FREQ] / (3.f * (float)BATCH * 257.f * 257.f);
    float Nt = (float)BATCH * 512.f * 512.f;
    float l1 = acc[IDX_L1] / Nt;
    float mse = acc[IDX_MSE] / Nt;

    out[0] = wv[0] * l1 + wv[1] * (1.f - ms) + wv[2] * gl + wv[3] * fl + wv[5] * mse;
}

// ---------------------------------------------------------------------------
extern "C" void kernel_launch(void* const* d_in, const int* in_sizes, int n_in,
                              void* d_out, int out_size, void* d_ws, size_t ws_size,
                              hipStream_t stream) {
    (void)in_sizes; (void)n_in; (void)out_size; (void)ws_size;
    const float* pred = (const float*)d_in[0];
    const float* targ = (const float*)d_in[1];
    const float* lw   = (const float*)d_in[2];
    const float* temp = (const float*)d_in[3];
    float* out = (float*)d_out;
    float* W   = (float*)d_ws;

    const int B = BATCH;
    // workspace layout (float offsets)
    size_t off = ACC_FLOATS;
    size_t Doff[5];
    for (int s = 0; s < 5; s++) { Doff[s] = off; int Hs = H0IMG >> s; off += (size_t)B * Hs * Hs; }
    size_t Xoff[5] = {0}, Yoff[5] = {0};
    for (int s = 1; s < 5; s++) {
        int Hs = H0IMG >> s;
        Xoff[s] = off; off += (size_t)B * Hs * Hs;
        Yoff[s] = off; off += (size_t)B * Hs * Hs;
    }
    size_t LOo = off; off += (size_t)B * 257 * 512;
    size_t HIo = off; off += (size_t)B * 257 * 512;
    size_t To  = off; off += (size_t)5 * B * 512 * 502;   // h-filtered planes (scale 0 max)

    // 1) zero accumulators
    k_zero<<<1, ACC_FLOATS, 0, stream>>>(W);

    // 2) d = pred - target, L1/MSE sums
    size_t N0 = (size_t)B * H0IMG * H0IMG;
    k_diff<<<2048, 256, 0, stream>>>(pred, targ, W + Doff[0], W, N0);

    // 3) MS-SSIM pyramid (WMMA conv kernels)
    for (int s = 0; s < 5; s++) {
        int Hs = H0IMG >> s, Ws = Hs, Wo = Ws - 10;
        const float* Xs = (s == 0) ? pred : (W + Xoff[s]);
        const float* Ys = (s == 0) ? targ : (W + Yoff[s]);
        int ct = (Wo + 15) / 16;
        int rtH = Hs / 16;
        int wavesH = ct * rtH * B;
        k_ssim_h<<<(wavesH + 7) / 8, 256, 0, stream>>>(Xs, Ys, W + To, Hs, Ws);
        int rtV = (Hs - 10 + 15) / 16;
        int wavesV = ct * rtV * B;
        k_ssim_v<<<(wavesV + 7) / 8, 256, 0, stream>>>(W + To, W, Hs, Wo, s);
        if (s < 4) {
            int Ho = Hs >> 1;
            size_t np = (size_t)B * Ho * Ho;
            int pb = (int)((np + 255) / 256);
            k_pool2<<<pb, 256, 0, stream>>>(Xs, W + Xoff[s + 1], Hs, Ws);
            k_pool2<<<pb, 256, 0, stream>>>(Ys, W + Yoff[s + 1], Hs, Ws);
        }
    }

    // 4) gradient loss (on pooled difference planes)
    for (int s = 0; s < 5; s++) {
        int Hs = H0IMG >> s;
        if (s > 0) {
            size_t np = (size_t)B * Hs * Hs;
            k_pool2<<<(int)((np + 255) / 256), 256, 0, stream>>>(
                W + Doff[s - 1], W + Doff[s], Hs << 1, Hs << 1);
        }
        k_grad<<<1024, 256, 0, stream>>>(W + Doff[s], Hs, Hs, s, W);
    }

    // 5) frequency loss: db2 DWT of the difference + Huber reduction
    size_t nv = (size_t)B * 257 * 512;
    k_dwt_v<<<(int)((nv + 255) / 256), 256, 0, stream>>>(W + Doff[0], W + LOo, W + HIo);
    size_t nh = (size_t)B * 257 * 257;
    k_dwt_h<<<(int)((nh + 255) / 256), 256, 0, stream>>>(W + LOo, W + HIo, W);

    // 6) combine with softmax weights
    k_final<<<1, 32, 0, stream>>>(W, lw, temp, out);
}
